// Recognition_87591563035157
// MI455X (gfx1250) — matrix operand order repaired and checked by reading, not compiled, not run on hardware
//
#include <hip/hip_runtime.h>
#include <hip/hip_bf16.h>

typedef __attribute__((ext_vector_type(16))) _Float16 v16h;
typedef __attribute__((ext_vector_type(8)))  float    v8f;
typedef __attribute__((ext_vector_type(4)))  int      v4i_;
typedef v4i_ __attribute__((address_space(1))) gv4;   // global int4
typedef v4i_ __attribute__((address_space(3))) lv4;   // LDS int4

// ---- problem constants ----
#define NN 32
#define CC 64
#define TD 400
#define VV 27
#define SS 2

// ================= CDNA5 async-copy helpers (guarded; fallback = plain copy) ==========
__device__ __forceinline__ void async_g2l_b128(const void* g, void* l) {
#if __has_builtin(__builtin_amdgcn_global_load_async_to_lds_b128)
  __builtin_amdgcn_global_load_async_to_lds_b128((gv4*)g, (lv4*)l, 0, 0);
#else
  *(v4i_*)l = *(const v4i_*)g;
#endif
}
__device__ __forceinline__ void async_l2g_b128(void* g, const void* l) {
#if __has_builtin(__builtin_amdgcn_global_store_async_from_lds_b128)
  __builtin_amdgcn_global_store_async_from_lds_b128((gv4*)g, (lv4*)l, 0, 0);
#else
  *(v4i_*)g = *(const v4i_*)l;
#endif
}
__device__ __forceinline__ void wait_async0() {
#if __has_builtin(__builtin_amdgcn_s_wait_asynccnt)
  __builtin_amdgcn_s_wait_asynccnt(0);
#else
  asm volatile("s_wait_asynccnt 0x0" ::: "memory");
#endif
}

// ================= WMMA fragment helpers (wave32, 16x16x32 f16) =================
__device__ __forceinline__ v16h frag_a16(const _Float16* base, int ldm, int lane) {
  v16h f;
  int m  = lane & 15;
  int kh = (lane >> 4) << 4;              // 0 or 16
  const _Float16* p = base + m * ldm;
#pragma unroll
  for (int r = 0; r < 8; ++r) {
    int k = kh + (r << 1);
    f[2 * r]     = p[k];
    f[2 * r + 1] = p[k + 1];
  }
  return f;
}
__device__ __forceinline__ v16h frag_b16(const _Float16* base, int ldn, int lane) {
  v16h f;
  int n  = lane & 15;
  int kb = (lane >> 4) << 4;              // 0 or 16
  const _Float16* p = base + n * ldn + kb;
#pragma unroll
  for (int r = 0; r < 8; ++r) {
    f[2 * r]     = p[2 * r];
    f[2 * r + 1] = p[2 * r + 1];
  }
  return f;
}
__device__ __forceinline__ v8f wmma16(v16h a, v16h b, v8f c) {
  return __builtin_amdgcn_wmma_f32_16x16x32_f16(false, a, false, b, (short)0, c, false, false);
}
__device__ __forceinline__ float lrelu(float x) { return x >= 0.f ? x : 0.1f * x; }

// ================= k0: zero att_raw =================
__global__ void k0_zero(float* att_raw) {
  int i = blockIdx.x * 256 + threadIdx.x;
  if (i < NN * SS * VV * VV) att_raw[i] = 0.f;
}

// ================= k1: qk = w_in*(x+pe)+b_in, accumulate att partials =================
#define TT1 16
#define PP1 (TT1 * VV)   // 432 = 27 tiles of 16
__global__ __launch_bounds__(256) void k1_qk_att(
    const float* __restrict__ x, const float* __restrict__ pe,
    const float* __restrict__ w_in, const float* __restrict__ b_in,
    float* __restrict__ att_raw) {
  extern __shared__ _Float16 lds[];
  _Float16* yT = lds;                 // [432][66]  pos-major (x+pe), c contiguous
  _Float16* wA = yT + 432 * 66;       // [64][66]
  _Float16* qT = wA + 64 * 66;        // [2][32][258]
  _Float16* kT = qT + 2 * 32 * 258;   // [2][32][258]
  const int tid  = threadIdx.x, lane = tid & 31, wid = tid >> 5;
  const int n = blockIdx.x / (TD / TT1), tb = blockIdx.x % (TD / TT1), t0 = tb * TT1;

  for (int e = tid; e < 64 * 64; e += 256) {
    int o = e >> 6, c = e & 63;
    wA[o * 66 + c] = (_Float16)w_in[o * 64 + c];
  }
  for (int e = tid; e < 2 * 32 * 258; e += 256) { qT[e] = (_Float16)0.f; kT[e] = (_Float16)0.f; }
  const float* xb = x + (size_t)n * CC * TD * VV;
  for (int c = 0; c < CC; ++c) {
    const float* xc = xb + (size_t)c * TD * VV + t0 * VV;
    const float* pc = pe + (size_t)c * TD * VV + t0 * VV;
    if (c + 1 < CC) {                          // global_prefetch_b8 of next channel row
      __builtin_prefetch(xc + TD * VV + tid, 0, 1);
      __builtin_prefetch(pc + TD * VV + tid, 0, 1);
    }
    for (int p = tid; p < PP1; p += 256) yT[p * 66 + c] = (_Float16)(xc[p] + pc[p]);
  }
  __syncthreads();

  // GEMM1: qk[o][pos] = w_in[o][c] * yT[pos][c], K=64
  for (int tile = wid; tile < 4 * 27; tile += 8) {
    int mt = tile / 27, nt = tile % 27;
    v8f acc = {};
#pragma unroll
    for (int kk = 0; kk < 64; kk += 32)
      acc = wmma16(frag_a16(wA + mt * 16 * 66 + kk, 66, lane),
                   frag_b16(yT + nt * 16 * 66 + kk, 66, lane), acc);
    int pos = nt * 16 + (lane & 15);
    int ttv = pos / VV, u = pos % VV;
    int mrow = (lane >> 4) << 3;
#pragma unroll
    for (int r = 0; r < 8; ++r) {
      int o = mt * 16 + mrow + r;
      float val = acc[r] + b_in[o];
      int s2 = o >> 4, ci = o & 15;
      _Float16* dst = (s2 < 2) ? (qT + s2 * 32 * 258) : (kT + (s2 - 2) * 32 * 258);
      dst[u * 258 + ttv * 16 + ci] = (_Float16)val;
    }
  }
  __syncthreads();

  // att partial: K = ci*TT1 = 256; one 16x16 tile per wave
  {
    int s = wid >> 2, mt = (wid >> 1) & 1, nt = wid & 1;
    const _Float16* qs = qT + s * 32 * 258;
    const _Float16* ks = kT + s * 32 * 258;
    v8f acc = {};
#pragma unroll
    for (int kk = 0; kk < 256; kk += 32)
      acc = wmma16(frag_a16(qs + mt * 16 * 258 + kk, 258, lane),
                   frag_b16(ks + nt * 16 * 258 + kk, 258, lane), acc);
    int v = nt * 16 + (lane & 15);
    int mrow = (lane >> 4) << 3;
#pragma unroll
    for (int r = 0; r < 8; ++r) {
      int u = mt * 16 + mrow + r;
      if (u < VV && v < VV)
        atomicAdd(att_raw + ((n * SS + s) * VV + u) * VV + v, acc[r]);
    }
  }
}

// ================= k2: finalize att -> f16 transposed [n][s][v][u] 32x32 padded =========
__global__ void k2_att(const float* __restrict__ att_raw, const float* __restrict__ alphas,
                       const float* __restrict__ att0, _Float16* __restrict__ att_h) {
  int idx = blockIdx.x * 256 + threadIdx.x;
  if (idx >= NN * SS * 32 * 32) return;
  int u = idx & 31, v = (idx >> 5) & 31, s = (idx >> 10) & 1, n = idx >> 11;
  float val = 0.f;
  if (u < VV && v < VV) {
    float raw = att_raw[((n * SS + s) * VV + u) * VV + v];
    val = tanhf(raw / (16.f * 400.f)) * alphas[s] + att0[(s * VV + u) * VV + v];
  }
  att_h[idx] = (_Float16)val;
}

// ================= k3: attention-apply + w_out(BN1,res) + w_ff(BN2,res) -> y5 (+f16 copy) ====
#define TT3 8
#define PP3 (TT3 * VV)     // 216
__global__ __launch_bounds__(256) void k3_fused(
    const float* __restrict__ x, const _Float16* __restrict__ att_h,
    const float* __restrict__ w_out, const float* __restrict__ b_out,
    const float* __restrict__ g1, const float* __restrict__ be1,
    const float* __restrict__ m1, const float* __restrict__ v1,
    const float* __restrict__ w_ff, const float* __restrict__ b_ff,
    const float* __restrict__ g2, const float* __restrict__ be2,
    const float* __restrict__ m2, const float* __restrict__ v2,
    float* __restrict__ y5, _Float16* __restrict__ y5h) {
  extern __shared__ _Float16 lds[];
  _Float16* xA   = lds;                  // [512][34]
  _Float16* y2   = xA + 512 * 34;        // [224][130]; reused as y5h staging [216][64] later
  _Float16* y4   = y2 + 224 * 130;       // [224][66]
  _Float16* wO   = y4 + 224 * 66;        // [64][130]
  _Float16* wF   = wO + 64 * 130;        // [64][66]
  _Float16* attL = wF + 64 * 66;         // [2][32][34]
  const int tid = threadIdx.x, lane = tid & 31, wid = tid >> 5;
  const int n = blockIdx.x / (TD / TT3), tb = blockIdx.x % (TD / TT3), t0 = tb * TT3;

  for (int e = tid; e < 64 * 128; e += 256) { int o = e >> 7, c = e & 127; wO[o * 130 + c] = (_Float16)w_out[o * 128 + c]; }
  for (int e = tid; e < 64 * 64;  e += 256) { int o = e >> 6, c = e & 63;  wF[o * 66 + c]  = (_Float16)w_ff[o * 64 + c]; }
  for (int e = tid; e < 2 * 32 * 32; e += 256) {
    int u = e & 31, v = (e >> 5) & 31, s = e >> 10;
    attL[(s * 32 + v) * 34 + u] = att_h[n * 2048 + e];
  }
  for (int e = tid; e < 512 * 34; e += 256) xA[e] = (_Float16)0.f;
  for (int e = tid; e < 8 * 130;  e += 256) y2[216 * 130 + e] = (_Float16)0.f;
  for (int e = tid; e < 8 * 66;   e += 256) y4[216 * 66 + e]  = (_Float16)0.f;
  __syncthreads();
  const float* xb = x + (size_t)n * CC * TD * VV + (size_t)t0 * VV;
  for (int c = 0; c < CC; ++c) {
    if (c + 1 < CC) __builtin_prefetch(xb + (size_t)(c + 1) * TD * VV + tid, 0, 1);
    for (int p = tid; p < PP3; p += 256) {
      int tt = p / VV, u = p % VV;
      xA[(c * TT3 + tt) * 34 + u] = (_Float16)xb[(size_t)c * TD * VV + p];
    }
  }
  __syncthreads();

  // GEMM-A: y2[s,c,t,v] = sum_u x[c,t,u]*att[s,u,v]; K=32
  for (int tile = wid; tile < 128; tile += 8) {
    int s = tile >> 6, mt = (tile >> 1) & 31, nt = tile & 1;
    v8f acc = {};
    acc = wmma16(frag_a16(xA + mt * 16 * 34, 34, lane),
                 frag_b16(attL + (s * 32 + nt * 16) * 34, 34, lane), acc);
    int v = nt * 16 + (lane & 15);
    if (v < VV) {
      int mrow = (lane >> 4) << 3;
#pragma unroll
      for (int r = 0; r < 8; ++r) {
        int m = mt * 16 + mrow + r, c = m >> 3, tt = m & 7;
        y2[(tt * VV + v) * 130 + s * 64 + c] = (_Float16)acc[r];
      }
    }
  }
  __syncthreads();

  // GEMM2: y3 = w_out * y2 (K=128) -> BN1 + x + lrelu -> y4[pos][o]
  for (int tile = wid; tile < 4 * 14; tile += 8) {
    int mt = tile / 14, nt = tile % 14;
    v8f acc = {};
#pragma unroll
    for (int kk = 0; kk < 128; kk += 32)
      acc = wmma16(frag_a16(wO + mt * 16 * 130 + kk, 130, lane),
                   frag_b16(y2 + nt * 16 * 130 + kk, 130, lane), acc);
    int p = nt * 16 + (lane & 15);
    if (p < PP3) {
      int tt = p / VV, v = p % VV, mrow = (lane >> 4) << 3;
#pragma unroll
      for (int r = 0; r < 8; ++r) {
        int o = mt * 16 + mrow + r;
        float f = acc[r] + b_out[o];
        f = (f - m1[o]) * (g1[o] * rsqrtf(v1[o] + 1e-5f)) + be1[o];
        f = lrelu(f + xb[(size_t)o * TD * VV + tt * VV + v]);
        y4[p * 66 + o] = (_Float16)f;
      }
    }
  }
  __syncthreads();

  // GEMM3: f = w_ff * y4 (K=64) -> BN2 + x + lrelu -> y5 (f32) + LDS staging (f16)
  _Float16* stg = y2;  // y2 is dead after GEMM2; reuse as packed [216][64] staging
  float* y5b = y5 + (size_t)n * CC * TD * VV + (size_t)t0 * VV;
  for (int tile = wid; tile < 4 * 14; tile += 8) {
    int mt = tile / 14, nt = tile % 14;
    v8f acc = {};
#pragma unroll
    for (int kk = 0; kk < 64; kk += 32)
      acc = wmma16(frag_a16(wF + mt * 16 * 66 + kk, 66, lane),
                   frag_b16(y4 + nt * 16 * 66 + kk, 66, lane), acc);
    int p = nt * 16 + (lane & 15);
    if (p < PP3) {
      int tt = p / VV, v = p % VV, mrow = (lane >> 4) << 3;
#pragma unroll
      for (int r = 0; r < 8; ++r) {
        int o = mt * 16 + mrow + r;
        float f = acc[r] + b_ff[o];
        f = (f - m2[o]) * (g2[o] * rsqrtf(v2[o] + 1e-5f)) + be2[o];
        f = lrelu(f + xb[(size_t)o * TD * VV + tt * VV + v]);
        y5b[(size_t)o * TD * VV + tt * VV + v] = f;
        stg[p * 64 + o] = (_Float16)f;
      }
    }
  }
  __syncthreads();
  // async writeback of the packed f16 copy: [pos][c] rows of 128B, contiguous
  {
    _Float16* yhb = y5h + ((size_t)n * TD * VV + (size_t)t0 * VV) * CC;
    for (int e = tid; e < PP3 * 8; e += 256)   // 1728 x 16B chunks
      async_l2g_b128(yhb + e * 8, stg + e * 8);
    wait_async0();
  }
}

// ================= k4: temporal conv (3 taps, K=64 each) + BN3 + res + lrelu ============
#define TT4 16
#define YLD 72            // LDS row pitch (halfs): 144B, 16B-aligned for B128 async
__global__ __launch_bounds__(256) void k4_conv(
    const float* __restrict__ y5, const _Float16* __restrict__ y5h,
    const float* __restrict__ w_t, const float* __restrict__ b_t,
    const float* __restrict__ g3, const float* __restrict__ be3,
    const float* __restrict__ m3, const float* __restrict__ v3,
    float* __restrict__ out) {
  extern __shared__ _Float16 lds[];
  _Float16* yB = lds;                 // [(16+2)*27=486][YLD]  halo rows, c contiguous
  _Float16* wT = yB + 486 * YLD;      // [3][64][66]
  const int tid = threadIdx.x, lane = tid & 31, wid = tid >> 5;
  const int n = blockIdx.x / (TD / TT4), tb = blockIdx.x % (TD / TT4), t0 = tb * TT4;

  // async-stage the f16 y5 copy with a 1-t halo on each side; zero-fill out of range
  {
    const _Float16* yh = y5h + (size_t)n * TD * VV * CC;
    int rowbase = t0 * VV - VV;
    for (int e = tid; e < 486 * 8; e += 256) {
      int h = e >> 3, j = e & 7;
      int grow = rowbase + h;
      _Float16* ldst = yB + h * YLD + j * 8;
      if (grow >= 0 && grow < TD * VV)
        async_g2l_b128(yh + ((size_t)grow * CC + j * 8), ldst);
      else {
        v4i_ z = {};
        *(v4i_*)ldst = z;
      }
    }
  }
  for (int e = tid; e < 3 * 64 * 64; e += 256) {
    int dt = e / 4096, rem = e & 4095, o = rem >> 6, c = rem & 63;
    wT[(dt * 64 + o) * 66 + c] = (_Float16)w_t[(o * 64 + c) * 3 + dt];
  }
  wait_async0();
  __syncthreads();

  const float* yb = y5 + (size_t)n * CC * TD * VV;
  float* ob = out + (size_t)n * CC * TD * VV + (size_t)t0 * VV;
  for (int tile = wid; tile < 4 * 27; tile += 8) {
    int mt = tile / 27, nt = tile % 27;
    v8f acc = {};
#pragma unroll
    for (int dt = 0; dt < 3; ++dt) {
      const _Float16* ab = wT + (dt * 64 + mt * 16) * 66;
      const _Float16* bb = yB + (nt * 16 + dt * VV) * YLD;  // row p+dt*27 == t+dt-1
#pragma unroll
      for (int kk = 0; kk < 64; kk += 32)
        acc = wmma16(frag_a16(ab + kk, 66, lane), frag_b16(bb + kk, YLD, lane), acc);
    }
    int p = nt * 16 + (lane & 15);
    int tt = p / VV, v = p % VV, mrow = (lane >> 4) << 3;
#pragma unroll
    for (int r = 0; r < 8; ++r) {
      int o = mt * 16 + mrow + r;
      float z = acc[r] + b_t[o];
      z = (z - m3[o]) * (g3[o] * rsqrtf(v3[o] + 1e-5f)) + be3[o];
      z = lrelu(z + yb[((size_t)o * TD + t0 + tt) * VV + v]);
      ob[(size_t)o * TD * VV + tt * VV + v] = z;
    }
  }
}

// ================= host =================
extern "C" void kernel_launch(void* const* d_in, const int* in_sizes, int n_in,
                              void* d_out, int out_size, void* d_ws, size_t ws_size,
                              hipStream_t stream) {
  const float* x      = (const float*)d_in[0];
  const float* pe     = (const float*)d_in[1];
  const float* w_in   = (const float*)d_in[2];
  const float* b_in   = (const float*)d_in[3];
  const float* alphas = (const float*)d_in[4];
  const float* att0   = (const float*)d_in[5];
  const float* w_out  = (const float*)d_in[6];
  const float* b_out  = (const float*)d_in[7];
  const float* bn1_g  = (const float*)d_in[8];
  const float* bn1_b  = (const float*)d_in[9];
  const float* bn1_m  = (const float*)d_in[10];
  const float* bn1_v  = (const float*)d_in[11];
  const float* w_ff   = (const float*)d_in[12];
  const float* b_ff   = (const float*)d_in[13];
  const float* bn2_g  = (const float*)d_in[14];
  const float* bn2_b  = (const float*)d_in[15];
  const float* bn2_m  = (const float*)d_in[16];
  const float* bn2_v  = (const float*)d_in[17];
  const float* w_t    = (const float*)d_in[18];
  const float* b_t    = (const float*)d_in[19];
  const float* bn3_g  = (const float*)d_in[20];
  const float* bn3_b  = (const float*)d_in[21];
  const float* bn3_m  = (const float*)d_in[22];
  const float* bn3_v  = (const float*)d_in[23];

  const size_t y5_bytes = (size_t)NN * CC * TD * VV * 4;               // 88.5 MB
  float*     att_raw = (float*)d_ws;                                   // 186,624 B
  _Float16*  att_h   = (_Float16*)((char*)d_ws + (256 << 10));
  float*     y5      = (float*)((char*)d_ws + (512 << 10));
  _Float16*  y5h     = (_Float16*)((char*)d_ws + (512 << 10) + y5_bytes);  // 44.2 MB

  const int natt = NN * SS * VV * VV;
  k0_zero<<<dim3((natt + 255) / 256), dim3(256), 0, stream>>>(att_raw);

  size_t lds1 = (size_t)(432 * 66 + 64 * 66 + 2 * (2 * 32 * 258)) * 2;
  k1_qk_att<<<dim3(NN * (TD / TT1)), dim3(256), lds1, stream>>>(x, pe, w_in, b_in, att_raw);

  k2_att<<<dim3((NN * SS * 32 * 32) / 256), dim3(256), 0, stream>>>(att_raw, alphas, att0, att_h);

  size_t lds3 = (size_t)(512 * 34 + 224 * 130 + 224 * 66 + 64 * 130 + 64 * 66 + 2 * 32 * 34) * 2;
  k3_fused<<<dim3(NN * (TD / TT3)), dim3(256), lds3, stream>>>(
      x, att_h, w_out, b_out, bn1_g, bn1_b, bn1_m, bn1_v,
      w_ff, b_ff, bn2_g, bn2_b, bn2_m, bn2_v, y5, y5h);

  size_t lds4 = (size_t)(486 * YLD + 3 * 64 * 66) * 2;
  k4_conv<<<dim3(NN * (TD / TT4)), dim3(256), lds4, stream>>>(
      y5, y5h, w_t, b_t, bn3_g, bn3_b, bn3_m, bn3_v, (float*)d_out);
}
